// MultiHeadAttention_2688649527861
// MI455X (gfx1250) — matrix operand order repaired and checked by reading, not compiled, not run on hardware
//
#include <hip/hip_runtime.h>

// ---------------------------------------------------------------------------
// Multi-head attention for MI455X (gfx1250), bf16 WMMA pipeline + TDM staging.
//   B=2, S=2048, E=1024, H=16, D=64.  Matmuls via v_wmma_f32_16x16x32_bf16.
// ---------------------------------------------------------------------------

typedef __attribute__((ext_vector_type(16))) __bf16 v16bf;
typedef __attribute__((ext_vector_type(8)))  float  v8f;
typedef unsigned v4u __attribute__((ext_vector_type(4)));
typedef int      v4i __attribute__((ext_vector_type(4)));
typedef int      v8i __attribute__((ext_vector_type(8)));

union FragBF { v16bf v; unsigned u[8]; };

#define S_LEN 2048
#define E_DIM 1024
#define H_NUM 16
#define D_DIM 64
#define B_NUM 2
#define M_ROWS (B_NUM * S_LEN)   // 4096

#if defined(__has_builtin)
#if __has_builtin(__builtin_amdgcn_tensor_load_to_lds)
#define USE_TDM 1
#endif
#if __has_builtin(__builtin_amdgcn_s_wait_tensorcnt)
#define HAVE_TENSORCNT 1
#endif
#if __has_builtin(__builtin_amdgcn_permlane16)
#define HAVE_PERMLANE16 1
#endif
#endif

__device__ __forceinline__ void wait_tensor2() {
#if defined(HAVE_TENSORCNT)
    __builtin_amdgcn_s_wait_tensorcnt(2);
#else
    asm volatile("s_wait_tensorcnt 0x2" ::: "memory");
#endif
}
__device__ __forceinline__ void wait_tensor0() {
#if defined(HAVE_TENSORCNT)
    __builtin_amdgcn_s_wait_tensorcnt(0);
#else
    asm volatile("s_wait_tensorcnt 0x0" ::: "memory");
#endif
}

// xor-butterfly permute within 16-lane groups: pure VALU (v_permlane16_b32),
// avoids ds_bpermute LDS traffic.  SEL0/SEL1 = packed 4-bit lane selectors
// (lane i reads lane i^MASK of its group).
template <unsigned SEL0, unsigned SEL1, int MASK>
__device__ __forceinline__ float xor16(float x) {
#if defined(HAVE_PERMLANE16)
    unsigned u = __builtin_bit_cast(unsigned, x);
    u = __builtin_amdgcn_permlane16(u, u, SEL0, SEL1, false, false);
    return __builtin_bit_cast(float, u);
#else
    return __shfl_xor(x, MASK, 32);
#endif
}
__device__ __forceinline__ float rowmax16(float x) {
    x = fmaxf(x, xor16<0x67452301u, 0xEFCDAB89u, 1>(x));   // i^1
    x = fmaxf(x, xor16<0x54761032u, 0xDCFE98BAu, 2>(x));   // i^2
    x = fmaxf(x, xor16<0x32107654u, 0xBA98FEDCu, 4>(x));   // i^4
    x = fmaxf(x, xor16<0xFEDCBA98u, 0x76543210u, 8>(x));   // i^8
    return x;
}

__device__ __forceinline__ unsigned short f2bf(float x) {
    unsigned u = __builtin_bit_cast(unsigned, x);
    u = (u + 0x7FFFu + ((u >> 16) & 1u)) >> 16;
    return (unsigned short)u;
}
__device__ __forceinline__ unsigned pack2bf(float a, float b) {
    return (unsigned)f2bf(a) | ((unsigned)f2bf(b) << 16);
}

// A-matrix (16x32 bf16) per-VGPR k offset, per ISA 7.12.2
__device__ __forceinline__ int a_koff(int v, int half) {
    return ((v < 4) ? 0 : 16) + half * 8 + (v & 3) * 2;
}
// B-matrix (32x16 bf16) per-VGPR k offset, per ISA 7.12.4/5 striping
__device__ __forceinline__ int b_koff(int v, int half) {
    return ((v < 4) ? 0 : 16) + half * 8 + (v & 3) * 2;
}

__device__ __forceinline__ v8f wmma_bf16(const FragBF& a, const FragBF& b, v8f c) {
    return __builtin_amdgcn_wmma_f32_16x16x32_bf16(
        false, a.v, false, b.v, (short)0, c, false, false);
}

// ---------------------------------------------------------------------------
// TDM: async-load a dense 2D bf16 tile (64 cols x 32 rows, row stride 64)
// from global into LDS.  One TENSOR_LOAD_TO_LDS per tile, TENSORcnt-tracked.
// Descriptor per CDNA5 ISA ch.8 (D# groups); clang-23 6-arg builtin form:
//   (uint32x4 g0, int32x8 g1, int32x4 g2, int32x4 g3, int32x8, i32 cpol)
// ---------------------------------------------------------------------------
__device__ __forceinline__ void tdm_load_tile_64x32(const unsigned short* gsrc,
                                                    unsigned short* ldst) {
#if defined(USE_TDM)
    const unsigned long long ga = (unsigned long long)(size_t)gsrc;
    const unsigned galo = __builtin_amdgcn_readfirstlane((unsigned)ga);
    const unsigned gahi = __builtin_amdgcn_readfirstlane((unsigned)(ga >> 32));
    const unsigned loff = __builtin_amdgcn_readfirstlane((unsigned)(size_t)ldst);
    // group0: count=1 | lds_addr | global_addr[56:0] | type=2
    v4u g0 = { 1u, loff, galo, (gahi & 0x01FFFFFFu) | (2u << 30) };
    // group1: data_size=2B; tensor_dim0=64; tensor_dim1=32; tile_dim0=64;
    //         tile_dim1=32; tensor_dim0_stride=64; no padding/iterate/barrier
    v8i g1 = { (int)(1u << 16),        // workgroup_mask=0, data_size=1(2B)
               (int)(64u << 16),       // atomic_barrier=0 | tensor_dim0[15:0]=64
               (int)(32u << 16),       // tensor_dim0[31:16]=0 | tensor_dim1[15:0]=32
               (int)(64u << 16),       // tensor_dim1[31:16]=0 | tile_dim0=64
               32,                     // tile_dim1=32 | tile_dim2=0
               64,                     // tensor_dim0_stride[31:0]=64
               0, 0 };                 // stride hi / tensor_dim1_stride = 0
    v4i gz4 = { 0, 0, 0, 0 };          // groups 2/3 unused (2D tile)
    v8i gz8 = { 0, 0, 0, 0, 0, 0, 0, 0 };
    __builtin_amdgcn_tensor_load_to_lds(g0, g1, gz4, gz4, gz8, 0);
#else
    // Fallback: synchronous wave copy (lane = row, 8 x 16B per row).
    const int lane = threadIdx.x & 31;
    const uint4* src = (const uint4*)(gsrc + (size_t)lane * 64);
    uint4* dst = (uint4*)(ldst + (size_t)lane * 64);
    #pragma unroll
    for (int c = 0; c < 8; ++c) dst[c] = src[c];
#endif
}

// ---------------------------------------------------------------------------
// GEMM: Out[m,n] = scale * sum_k A[m,k] * W[n,k]   (K = N = 1024, M = 4096)
// Double-buffered LDS (ping-pong), one barrier per 32-K chunk.
// 256 threads = 8 waves in 2(M) x 4(N); wave tile 64x32 = 4x2 WMMA accums.
// ---------------------------------------------------------------------------
template <bool A_BF16, bool OUT_QKV>
__global__ __launch_bounds__(256) void gemm_wT_kernel(
    const void* __restrict__ Araw, const float* __restrict__ W,
    void* __restrict__ Out, float scale)
{
    constexpr int K = E_DIM, N = E_DIM;
    constexpr int BM = 128, BN = 128, BK = 32;
    __shared__ __align__(16) unsigned as_u[2][BM * BK / 2]; // 2 x 8 KB
    __shared__ __align__(16) unsigned bs_u[2][BN * BK / 2]; // 2 x 8 KB

    const int tid  = threadIdx.x;
    const int lane = tid & 31;
    const int wave = tid >> 5;
    const int half = lane >> 4;
    const int lrow = lane & 15;
    const int wm   = wave >> 2;   // 0..1
    const int wn   = wave & 3;    // 0..3
    const int m0   = blockIdx.x * BM;
    const int n0   = blockIdx.y * BN;

    v8f acc[4][2];
    #pragma unroll
    for (int mt = 0; mt < 4; ++mt)
        #pragma unroll
        for (int nt = 0; nt < 2; ++nt) { v8f z = {}; acc[mt][nt] = z; }

    int ak[8], bk[8];
    #pragma unroll
    for (int v = 0; v < 8; ++v) { ak[v] = a_koff(v, half) >> 1; bk[v] = b_koff(v, half) >> 1; }

    auto stage = [&](int buf, int k0) {
        if constexpr (!A_BF16) {
            const float* A = (const float*)Araw;
            #pragma unroll
            for (int it = 0; it < 4; ++it) {
                int i   = tid + it * 256;            // 1024 float4
                int row = i >> 3, c4 = i & 7;
                const float4 val = *(const float4*)&A[(size_t)(m0 + row) * K + k0 + c4 * 4];
                as_u[buf][row * 16 + c4 * 2 + 0] = pack2bf(val.x, val.y);
                as_u[buf][row * 16 + c4 * 2 + 1] = pack2bf(val.z, val.w);
            }
        } else {
            const unsigned short* A = (const unsigned short*)Araw;
            #pragma unroll
            for (int it = 0; it < 2; ++it) {
                int i   = tid + it * 256;            // 512 uint4
                int row = i >> 2, c8 = i & 3;
                const uint4 val = *(const uint4*)&A[(size_t)(m0 + row) * K + k0 + c8 * 8];
                as_u[buf][row * 16 + c8 * 4 + 0] = val.x;
                as_u[buf][row * 16 + c8 * 4 + 1] = val.y;
                as_u[buf][row * 16 + c8 * 4 + 2] = val.z;
                as_u[buf][row * 16 + c8 * 4 + 3] = val.w;
            }
        }
        #pragma unroll
        for (int it = 0; it < 4; ++it) {
            int i   = tid + it * 256;
            int row = i >> 3, c4 = i & 7;
            const float4 wv = *(const float4*)&W[(size_t)(n0 + row) * K + k0 + c4 * 4];
            bs_u[buf][row * 16 + c4 * 2 + 0] = pack2bf(wv.x, wv.y);
            bs_u[buf][row * 16 + c4 * 2 + 1] = pack2bf(wv.z, wv.w);
        }
    };

    stage(0, 0);
    __syncthreads();

    for (int kc = 0; kc < K / BK; ++kc) {
        const int cur = kc & 1;
        if (kc + 1 < K / BK) stage(cur ^ 1, (kc + 1) * BK);

        FragBF afr[4], bfr[2];
        #pragma unroll
        for (int mt = 0; mt < 4; ++mt) {
            int r = wm * 64 + mt * 16 + lrow;
            #pragma unroll
            for (int v = 0; v < 8; ++v) afr[mt].u[v] = as_u[cur][r * 16 + ak[v]];
        }
        #pragma unroll
        for (int nt = 0; nt < 2; ++nt) {
            int r = wn * 32 + nt * 16 + lrow;
            #pragma unroll
            for (int v = 0; v < 8; ++v) bfr[nt].u[v] = bs_u[cur][r * 16 + bk[v]];
        }
        #pragma unroll
        for (int mt = 0; mt < 4; ++mt)
            #pragma unroll
            for (int nt = 0; nt < 2; ++nt)
                acc[mt][nt] = wmma_bf16(afr[mt], bfr[nt], acc[mt][nt]);
        __syncthreads();
    }

    #pragma unroll
    for (int mt = 0; mt < 4; ++mt) {
        #pragma unroll
        for (int nt = 0; nt < 2; ++nt) {
            #pragma unroll
            for (int v = 0; v < 8; ++v) {
                const int m = m0 + wm * 64 + mt * 16 + v + half * 8;
                const int n = n0 + wn * 32 + nt * 16 + lrow;
                const float val = acc[mt][nt][v] * scale;
                if constexpr (OUT_QKV) {
                    const int b = m >> 11, s = m & (S_LEN - 1);
                    const int head = n >> 6, d = n & (D_DIM - 1);
                    ((unsigned short*)Out)[((((size_t)b * H_NUM + head) * S_LEN + s) << 6) + d] = f2bf(val);
                } else {
                    ((float*)Out)[(size_t)m * N + n] = val;
                }
            }
        }
    }
}

// ---------------------------------------------------------------------------
// Flash attention: one wave per 16-query tile; K/V blocks (32 keys) staged to
// per-wave LDS double buffers via TDM, overlapped one block ahead.
// Row sums via WMMA against an all-ones fragment; row max via v_permlane16.
// No barriers: TDM + LDS are in-order per wave.
// ---------------------------------------------------------------------------
__global__ __launch_bounds__(128) void attn_kernel(
    const unsigned short* __restrict__ Qb,
    const unsigned short* __restrict__ Kb,
    const unsigned short* __restrict__ Vb,
    unsigned short* __restrict__ Ctx)
{
    __shared__ __align__(16) unsigned short kbuf[4][2][32 * 64]; // 32 KB
    __shared__ __align__(16) unsigned short vbuf[4][2][32 * 64]; // 32 KB
    __shared__ __align__(16) unsigned short pbuf[4][16 * 32];    //  4 KB

    const int tid  = threadIdx.x;
    const int lane = tid & 31;
    const int wave = tid >> 5;
    const int half = lane >> 4;
    const int lrow = lane & 15;
    const int bh   = blockIdx.y;                  // b*H + head
    const int q0   = (blockIdx.x * 4 + wave) * 16;
    const size_t base = (size_t)bh * S_LEN * D_DIM;
    const unsigned short* qp = Qb + base;
    const unsigned short* kp = Kb + base;
    const unsigned short* vp = Vb + base;
    unsigned short* pl = &pbuf[wave][0];

    int akv[8], bkv[8];
    #pragma unroll
    for (int v = 0; v < 8; ++v) { akv[v] = a_koff(v, half); bkv[v] = b_koff(v, half); }

    FragBF ones;
    #pragma unroll
    for (int v = 0; v < 8; ++v) ones.u[v] = 0x3F803F80u;  // bf16 1.0 pair

    // Q fragments: A matrix 16x32 over d (two chunks of 32)
    FragBF qf[2];
    #pragma unroll
    for (int c = 0; c < 2; ++c)
        #pragma unroll
        for (int v = 0; v < 8; ++v)
            qf[c].u[v] = *(const unsigned*)&qp[(size_t)(q0 + lrow) * D_DIM + c * 32 + akv[v]];

    v8f ctx[4];
    #pragma unroll
    for (int nt = 0; nt < 4; ++nt) { v8f z = {}; ctx[nt] = z; }
    float mrow[8], lsum[8];
    #pragma unroll
    for (int v = 0; v < 8; ++v) { mrow[v] = -1e30f; lsum[v] = 0.0f; }

    constexpr int NB = S_LEN / 32;   // 64 key blocks

    // Prologue: stage block 0 into buffer 0.
    tdm_load_tile_64x32(kp, &kbuf[wave][0][0]);
    tdm_load_tile_64x32(vp, &vbuf[wave][0][0]);

    for (int i = 0; i < NB; ++i) {
        const int cur = i & 1;
        if (i + 1 < NB) {
            tdm_load_tile_64x32(kp + (size_t)(i + 1) * 32 * D_DIM, &kbuf[wave][cur ^ 1][0]);
            tdm_load_tile_64x32(vp + (size_t)(i + 1) * 32 * D_DIM, &vbuf[wave][cur ^ 1][0]);
            wait_tensor2();          // block i's two tiles have landed
        } else {
            wait_tensor0();
        }
        const unsigned short* kl = &kbuf[wave][cur][0];
        const unsigned short* vl = &vbuf[wave][cur][0];

        // ---- scores: 16 queries x 32 keys ----
        v8f sc[2];
        { v8f z = {}; sc[0] = z; sc[1] = z; }
        #pragma unroll
        for (int nt = 0; nt < 2; ++nt) {
            #pragma unroll
            for (int c = 0; c < 2; ++c) {
                FragBF kf;  // B matrix: rows = d (k-dim), cols = key
                #pragma unroll
                for (int v = 0; v < 8; ++v)
                    kf.u[v] = *(const unsigned*)&kl[(nt * 16 + lrow) * D_DIM + c * 32 + akv[v]];
                sc[nt] = wmma_bf16(qf[c], kf, sc[nt]);
            }
        }

        // ---- online softmax: permlane16 max, WMMA row-sum ----
        float cscale[8];
        #pragma unroll
        for (int v = 0; v < 8; ++v) {
            float s0 = sc[0][v], s1 = sc[1][v];
            const float mx = rowmax16(fmaxf(s0, s1));
            const float nm = fmaxf(mrow[v], mx);
            const float p0 = __expf(s0 - nm);
            const float p1 = __expf(s1 - nm);
            cscale[v] = __expf(mrow[v] - nm);
            mrow[v]   = nm;
            const int m = v + half * 8;       // C-layout row
            pl[m * 32 + lrow]      = f2bf(p0);
            pl[m * 32 + 16 + lrow] = f2bf(p1);
        }
        #pragma unroll
        for (int nt = 0; nt < 4; ++nt)
            #pragma unroll
            for (int v = 0; v < 8; ++v) ctx[nt][v] *= cscale[v];

        // ---- P fragment from LDS (in-order per wave) ----
        FragBF pf;
        #pragma unroll
        for (int v = 0; v < 8; ++v)
            pf.u[v] = *(const unsigned*)&pl[lrow * 32 + akv[v]];

        // Row sums of P via WMMA vs all-ones: C row pattern matches stats.
        v8f rs; { v8f z = {}; rs = z; }
        rs = wmma_bf16(pf, ones, rs);
        #pragma unroll
        for (int v = 0; v < 8; ++v) lsum[v] = lsum[v] * cscale[v] + rs[v];

        // ---- ctx += P x V ----
        #pragma unroll
        for (int nt = 0; nt < 4; ++nt) {
            FragBF vf;  // B matrix: rows = key (k-dim), cols = d
            #pragma unroll
            for (int v = 0; v < 8; ++v) {
                const unsigned lo = vl[bkv[v] * D_DIM + nt * 16 + lrow];
                const unsigned hi = vl[(bkv[v] + 1) * D_DIM + nt * 16 + lrow];
                vf.u[v] = lo | (hi << 16);
            }
            ctx[nt] = wmma_bf16(pf, vf, ctx[nt]);
        }
    }

    // ---- normalize + write ctx bf16 [B*S, E] ----
    const int b = bh >> 4, head = bh & (H_NUM - 1);
    #pragma unroll
    for (int v = 0; v < 8; ++v) {
        const float inv = 1.0f / lsum[v];
        const int s = q0 + v + half * 8;
        const size_t rowbase = ((size_t)b * S_LEN + s) * E_DIM + head * D_DIM;
        #pragma unroll
        for (int nt = 0; nt < 4; ++nt)
            Ctx[rowbase + nt * 16 + lrow] = f2bf(ctx[nt][v] * inv);
    }
}

// ---------------------------------------------------------------------------
extern "C" void kernel_launch(void* const* d_in, const int* in_sizes, int n_in,
                              void* d_out, int out_size, void* d_ws, size_t ws_size,
                              hipStream_t stream) {
    const float* h  = (const float*)d_in[0];
    const float* Wq = (const float*)d_in[1];
    const float* Wk = (const float*)d_in[2];
    const float* Wv = (const float*)d_in[3];
    const float* Wo = (const float*)d_in[4];
    float* out = (float*)d_out;

    const size_t elems = (size_t)M_ROWS * E_DIM;      // 4M bf16 each
    unsigned short* qb  = (unsigned short*)d_ws;
    unsigned short* kb  = qb + elems;
    unsigned short* vb  = kb + elems;
    unsigned short* ctx = vb + elems;

    const dim3 gg(M_ROWS / 128, E_DIM / 128);         // (32, 8)
    const float qscale = 0.125f;                      // 1/sqrt(64)

    gemm_wT_kernel<false, true><<<gg, 256, 0, stream>>>(h, Wq, qb, qscale);
    gemm_wT_kernel<false, true><<<gg, 256, 0, stream>>>(h, Wk, kb, 1.0f);
    gemm_wT_kernel<false, true><<<gg, 256, 0, stream>>>(h, Wv, vb, 1.0f);

    attn_kernel<<<dim3(S_LEN / 16 / 4, B_NUM * H_NUM), 128, 0, stream>>>(qb, kb, vb, ctx);

    gemm_wT_kernel<true, false><<<gg, 256, 0, stream>>>(ctx, Wo, out, 1.0f);
}
